// Point_model_62182536511639
// MI455X (gfx1250) — matrix operand order repaired and checked by reading, not compiled, not run on hardware
//
#include <hip/hip_runtime.h>
#include <hip/hip_bf16.h>

#define BATCH 8
#define NPTS0 2048
#define KNB   32

typedef unsigned short u16;
typedef __bf16 bf16_t;
typedef bf16_t v16bf __attribute__((ext_vector_type(16)));
typedef float  v8f   __attribute__((ext_vector_type(8)));
typedef u16    u16x8 __attribute__((ext_vector_type(8)));

union FragAB { u16x8 h[2]; v16bf v; };

__device__ __forceinline__ u16 f2bf(float f) {
  union { float f; unsigned u; } x; x.f = f;
  unsigned u = x.u;
  if ((u & 0x7fffffffu) > 0x7f800000u) return (u16)((u >> 16) | 0x40u); // quiet NaN
  return (u16)((u + 0x7fffu + ((u >> 16) & 1u)) >> 16);                 // RNE
}
__device__ __forceinline__ float bf2f(u16 h) {
  union { unsigned u; float f; } x; x.u = ((unsigned)h) << 16; return x.f;
}

// ---------------------------------------------------------------------------
// transpose (B,3,N) -> (B,N,3)
__global__ void k_tin(const float* __restrict__ x, float* __restrict__ o, int N, long total) {
  long i = (long)blockIdx.x * blockDim.x + threadIdx.x;
  if (i >= total) return;
  int c = (int)(i % 3);
  long bn = i / 3;
  int n = (int)(bn % N);
  int b = (int)(bn / N);
  o[i] = x[((size_t)b * 3 + c) * N + n];
}

// transpose (B,N,C) -> (B,C,N)
__global__ void k_tout(const float* __restrict__ x, float* __restrict__ o, int N, int C, long total) {
  long i = (long)blockIdx.x * blockDim.x + threadIdx.x;
  if (i >= total) return;
  int n = (int)(i % N);
  long bc = i / N;
  int c = (int)(bc % C);
  int b = (int)(bc / C);
  o[i] = x[((size_t)b * N + n) * C + c];
}

// ---------------------------------------------------------------------------
// farthest point sampling: one block (256 thr) per batch; points+dist in LDS.
// wave32 shuffle argmax + one cross-wave combine -> 2 barriers per scan step.
__global__ __launch_bounds__(256) void k_fps(const float* __restrict__ xyz,
                                             int* __restrict__ fidx, int N, int npoint) {
  __shared__ float px[2048], py[2048], pz[2048], dist[2048];
  __shared__ float rmax[8];
  __shared__ int   rarg[8];
  __shared__ int   s_far;
  const int b = blockIdx.x;
  const int t = threadIdx.x;
  const int lane = t & 31;
  const int wv = t >> 5;
  const float* base = xyz + (size_t)b * N * 3;
  for (int i = t; i < N; i += 256) {
    px[i] = base[i * 3 + 0];
    py[i] = base[i * 3 + 1];
    pz[i] = base[i * 3 + 2];
    dist[i] = 1e10f;
  }
  __syncthreads();
  int far = 0;
  for (int it = 0; it < npoint; ++it) {
    if (t == 0) fidx[b * npoint + it] = far;
    const float fx = px[far], fy = py[far], fz = pz[far];
    float bm = -1.0f; int bi = 0x7fffffff;
    for (int i = t; i < N; i += 256) {
      float dx = px[i] - fx, dy = py[i] - fy, dz = pz[i] - fz;
      float d = dx * dx + dy * dy + dz * dz;
      float dd = fminf(dist[i], d);
      dist[i] = dd;
      if (dd > bm || (dd == bm && i < bi)) { bm = dd; bi = i; }
    }
    // wave32 argmax reduction
#pragma unroll
    for (int o = 16; o > 0; o >>= 1) {
      float om = __shfl_xor(bm, o, 32);
      int   oi = __shfl_xor(bi, o, 32);
      if (om > bm || (om == bm && oi < bi)) { bm = om; bi = oi; }
    }
    if (lane == 0) { rmax[wv] = bm; rarg[wv] = bi; }
    __syncthreads();
    if (t == 0) {
      float gm = rmax[0]; int gi = rarg[0];
#pragma unroll
      for (int w = 1; w < 8; ++w) {
        float om = rmax[w]; int oi = rarg[w];
        if (om > gm || (om == gm && oi < gi)) { gm = om; gi = oi; }
      }
      s_far = gi;
    }
    __syncthreads();
    far = s_far;
  }
}

// gather sampled centroids
__global__ void k_gather_xyz(const float* __restrict__ xyz, const int* __restrict__ fidx,
                             float* __restrict__ nxyz, int N, int S, int total) {
  int t = blockIdx.x * blockDim.x + threadIdx.x;
  if (t >= total) return;
  int b = t / S;
  int id = fidx[t];
  const float* p = xyz + ((size_t)b * N + id) * 3;
  float* o = nxyz + (size_t)t * 3;
  o[0] = p[0]; o[1] = p[1]; o[2] = p[2];
}

// ball query: first KNB indices within radius (pad with first hit)
__global__ void k_ball(const float* __restrict__ xyz, const float* __restrict__ nxyz,
                       int* __restrict__ idx, int N, int S, float r2, int total) {
  int t = blockIdx.x * blockDim.x + threadIdx.x;
  if (t >= total) return;
  int b = t / S;
  const float* q = nxyz + (size_t)t * 3;
  float qx = q[0], qy = q[1], qz = q[2];
  const float* base = xyz + (size_t)b * N * 3;
  int* o = idx + (size_t)t * KNB;
  int cnt = 0, first = -1;
  for (int i = 0; i < N && cnt < KNB; ++i) {
    float dx = base[i * 3] - qx, dy = base[i * 3 + 1] - qy, dz = base[i * 3 + 2] - qz;
    if (dx * dx + dy * dy + dz * dz <= r2) { if (first < 0) first = i; o[cnt++] = i; }
  }
  if (first < 0) first = N - 1;
  for (; cnt < KNB; ++cnt) o[cnt] = first;
}

// build grouped features [dxyz | points[idx] | 0-pad] as bf16, rows = B*S*KNB
__global__ void k_group(const float* __restrict__ xyz, const float* __restrict__ nxyz,
                        const int* __restrict__ idx, const float* __restrict__ pts,
                        u16* __restrict__ feats, int N, int S, int Cp, int lda, long total) {
  long i = (long)blockIdx.x * blockDim.x + threadIdx.x;
  if (i >= total) return;
  int c = (int)(i % lda);
  long r = i / lda;               // (b*S+s)*KNB + k
  long bs = r / KNB;
  int b = (int)(bs / S);
  float v = 0.0f;
  if (c < 3 + Cp) {
    int pid = idx[r];
    if (c < 3) v = xyz[((size_t)b * N + pid) * 3 + c] - nxyz[bs * 3 + c];
    else       v = pts[((size_t)b * N + pid) * Cp + (c - 3)];
  }
  feats[i] = f2bf(v);
}

// convert fp32 weights (cout x cin) to bf16 (cout x lda) with zero pad
__global__ void k_convw(const float* __restrict__ W, u16* __restrict__ Wb,
                        int cin, int lda, long total) {
  long i = (long)blockIdx.x * blockDim.x + threadIdx.x;
  if (i >= total) return;
  int c = (int)(i % lda);
  long o = i / lda;
  Wb[i] = (c < cin) ? f2bf(W[o * (long)cin + c]) : (u16)0;
}

__global__ void k_zero(float* __restrict__ p, int n) {
  int i = blockIdx.x * blockDim.x + threadIdx.x;
  if (i < n) p[i] = 0.0f;
}

// ---------------------------------------------------------------------------
// WMMA bf16 GEMM: Out[r,o] = sum_c A[r,c]*W[o,c] + bias[o]
// block = 8 waves; wave -> 32 rows x 64 cols (8 wmma tiles); grid (rows/256, cout/64)
// software-pipelined: register double-buffered A/B fragments (ping-pong, no copies)
// epilogue: bf16 store + per-channel sum/sumsq atomics for batchnorm stats.
__global__ __launch_bounds__(256) void k_gemm_wmma(
    const u16* __restrict__ A, const u16* __restrict__ W,
    const float* __restrict__ bias, u16* __restrict__ Out,
    float* __restrict__ stats, int rows, int lda, int cout) {
  const int lane = threadIdx.x & 31;
  const int wave = threadIdx.x >> 5;
  const int half = lane >> 4;
  const int l16  = lane & 15;
  const int row0 = blockIdx.x * 256 + wave * 32;
  const int nbase = blockIdx.y * 64;
  if (row0 >= rows) return;

  v8f acc[2][4] = {};

  const u16* arow0 = A + (size_t)(row0 + l16) * lda;
  const u16* arow1 = A + (size_t)(row0 + 16 + l16) * lda;
  const u16* wrow  = W + (size_t)(nbase + l16) * lda;

  FragAB aP[2], aQ[2], bP[4], bQ[4];

  auto loadA = [&](FragAB* a, int k0) {
    a[0].h[0] = *(const u16x8*)(arow0 + k0 + half * 8);        // K: half*8 .. +7
    a[0].h[1] = *(const u16x8*)(arow0 + k0 + 16 + half * 8);   // K: 16+half*8 .. +7
    a[1].h[0] = *(const u16x8*)(arow1 + k0 + half * 8);
    a[1].h[1] = *(const u16x8*)(arow1 + k0 + 16 + half * 8);
  };
  auto loadB = [&](FragAB* bfr, int k0) {
#pragma unroll
    for (int j = 0; j < 4; ++j) {
      const u16* wp = wrow + (size_t)j * 16 * lda + k0 + half * 8;
      bfr[j].h[0] = *(const u16x8*)(wp);                       // K: half*8 .. +7
      bfr[j].h[1] = *(const u16x8*)(wp + 16);                  // K: 16+half*8 .. +7
    }
  };
  auto compute = [&](FragAB* a, FragAB* bfr) {
#pragma unroll
    for (int m = 0; m < 2; ++m)
#pragma unroll
      for (int j = 0; j < 4; ++j)
        acc[m][j] = __builtin_amdgcn_wmma_f32_16x16x32_bf16(
            false, a[m].v, false, bfr[j].v, (short)0, acc[m][j], false, false);
  };

  loadA(aP, 0);
  loadB(bP, 0);
  int k0 = 32;
  for (;;) {
    if (k0 < lda) {
      loadA(aQ, k0);
      loadB(bQ, k0);
      if (k0 + 32 < lda) {                       // warm WGP$ one stage further
        __builtin_prefetch(arow0 + k0 + 32, 0, 1);
        __builtin_prefetch(arow1 + k0 + 32, 0, 1);
      }
      compute(aP, bP);
    } else { compute(aP, bP); break; }
    k0 += 32;
    if (k0 < lda) {
      loadA(aP, k0);
      loadB(bP, k0);
      if (k0 + 32 < lda) {
        __builtin_prefetch(arow0 + k0 + 32, 0, 1);
        __builtin_prefetch(arow1 + k0 + 32, 0, 1);
      }
      compute(aQ, bQ);
    } else { compute(aQ, bQ); break; }
    k0 += 32;
  }

#pragma unroll
  for (int j = 0; j < 4; ++j) {
    int col = nbase + j * 16 + l16;
    float bs = bias[col];
    float s = 0.0f, s2 = 0.0f;
#pragma unroll
    for (int m = 0; m < 2; ++m)
#pragma unroll
      for (int r = 0; r < 8; ++r) {
        int orow = row0 + m * 16 + r + half * 8;
        float v = acc[m][j][r] + bs;
        Out[(size_t)orow * cout + col] = f2bf(v);
        s += v; s2 += v * v;
      }
    atomicAdd(&stats[col], s);
    atomicAdd(&stats[cout + col], s2);
  }
}

// batchnorm (training-mode stats) + ReLU; bf16 in-place; optional fp32 copy out
__global__ void k_bnrelu(u16* __restrict__ act, const float* __restrict__ stats,
                         const float* __restrict__ gamma, const float* __restrict__ beta,
                         float invcnt, int cout, long total, float* __restrict__ outf) {
  long i = (long)blockIdx.x * blockDim.x + threadIdx.x;
  if (i >= total) return;
  int c = (int)(i % cout);
  float mean = stats[c] * invcnt;
  float var  = stats[cout + c] * invcnt - mean * mean;
  float sc = gamma[c] * rsqrtf(var + 1e-5f);
  float sh = beta[c] - mean * sc;
  float x = bf2f(act[i]) * sc + sh;
  x = fmaxf(x, 0.0f);
  act[i] = f2bf(x);
  if (outf) outf[i] = x;
}

// max over the K (=KNB) grouped samples: (B*S*K, C) bf16 -> (B*S, C) fp32
__global__ void k_maxpool(const u16* __restrict__ act, float* __restrict__ out,
                          int C, long total) {
  long i = (long)blockIdx.x * blockDim.x + threadIdx.x;
  if (i >= total) return;
  int c = (int)(i % C);
  long bs = i / C;
  const u16* p = act + (size_t)bs * KNB * C + c;
  float m = -1e30f;
  for (int k = 0; k < KNB; ++k) m = fmaxf(m, bf2f(p[(size_t)k * C]));
  out[i] = m;
}

// 3-nearest-neighbor + inverse-distance weights for feature propagation
__global__ void k_knn3(const float* __restrict__ xyz1, const float* __restrict__ xyz2,
                       int* __restrict__ nidx, float* __restrict__ nw,
                       int N1, int N2, int total) {
  int t = blockIdx.x * blockDim.x + threadIdx.x;
  if (t >= total) return;
  int b = t / N1;
  const float* q = xyz1 + (size_t)t * 3;
  float qx = q[0], qy = q[1], qz = q[2];
  const float* base = xyz2 + (size_t)b * N2 * 3;
  float d0 = 3.4e38f, d1 = 3.4e38f, d2 = 3.4e38f;
  int i0 = 0, i1 = 0, i2 = 0;
  for (int m = 0; m < N2; ++m) {
    float dx = base[m * 3] - qx, dy = base[m * 3 + 1] - qy, dz = base[m * 3 + 2] - qz;
    float d = dx * dx + dy * dy + dz * dz;
    if (d < d0)      { d2 = d1; i2 = i1; d1 = d0; i1 = i0; d0 = d; i0 = m; }
    else if (d < d1) { d2 = d1; i2 = i1; d1 = d;  i1 = m; }
    else if (d < d2) { d2 = d;  i2 = m; }
  }
  float w0 = 1.0f / (d0 + 1e-8f), w1 = 1.0f / (d1 + 1e-8f), w2 = 1.0f / (d2 + 1e-8f);
  float ws = w0 + w1 + w2;
  nidx[t * 3 + 0] = i0; nidx[t * 3 + 1] = i1; nidx[t * 3 + 2] = i2;
  nw[t * 3 + 0] = w0 / ws; nw[t * 3 + 1] = w1 / ws; nw[t * 3 + 2] = w2 / ws;
}

// build FP-layer input rows [points1 | interp(points2) | 0-pad] as bf16
__global__ void k_interp(const float* __restrict__ p1, const float* __restrict__ p2,
                         const int* __restrict__ nidx, const float* __restrict__ nw,
                         u16* __restrict__ feats, int N1, int N2, int C1, int C2,
                         int lda, long total) {
  long i = (long)blockIdx.x * blockDim.x + threadIdx.x;
  if (i >= total) return;
  int c = (int)(i % lda);
  long r = i / lda;               // b*N1 + n
  int b = (int)(r / N1);
  float v = 0.0f;
  if (c < C1) {
    v = p1[r * (long)C1 + c];
  } else if (c < C1 + C2) {
    int cc = c - C1;
    const int* id = nidx + r * 3;
    const float* w = nw + r * 3;
    const float* pb = p2 + (size_t)b * N2 * C2;
    v = w[0] * pb[(size_t)id[0] * C2 + cc] +
        w[1] * pb[(size_t)id[1] * C2 + cc] +
        w[2] * pb[(size_t)id[2] * C2 + cc];
  }
  feats[i] = f2bf(v);
}

// ---------------------------------------------------------------------------
extern "C" void kernel_launch(void* const* d_in, const int* in_sizes, int n_in,
                              void* d_out, int out_size, void* d_ws, size_t ws_size,
                              hipStream_t stream) {
  (void)in_sizes; (void)n_in; (void)out_size; (void)ws_size;

  // ---- parameter parsing (setup_inputs dict insertion order) ----
  struct MlpP { const float *W[3], *b[3], *g[3], *bt[3]; int n; };
  int pi = 1;  // d_in[0] = xyz
  auto read_mlp = [&](int nconv) {
    MlpP m; m.n = nconv;
    for (int i = 0; i < nconv; ++i) { m.W[i] = (const float*)d_in[pi++]; m.b[i]  = (const float*)d_in[pi++]; }
    for (int i = 0; i < nconv; ++i) { m.g[i] = (const float*)d_in[pi++]; m.bt[i] = (const float*)d_in[pi++]; }
    return m;
  };
  MlpP sa1 = read_mlp(3), sa2 = read_mlp(3), sa3 = read_mlp(3), sa4 = read_mlp(3);
  MlpP fp4 = read_mlp(2), fp3 = read_mlp(2), fp2 = read_mlp(2), fp1 = read_mlp(3);
  static const int sa1c[3] = {128, 128, 256},  sa2c[3] = {256, 256, 512};
  static const int sa3c[3] = {512, 512, 1024}, sa4c[3] = {1024, 1024, 2048};
  static const int fp4c[2] = {1024, 1024}, fp3c[2] = {512, 512};
  static const int fp2c[2] = {256, 256},   fp1c[3] = {128, 128, 128};

  // ---- workspace carve ----
  char* base = (char*)d_ws; size_t off = 0;
  auto carve = [&](size_t bytes) -> char* {
    char* r = base + off; off = (off + bytes + 255) & ~(size_t)255; return r;
  };
  u16*   actA  = (u16*)carve(134217728);   // 262144 x 256 bf16
  u16*   actB  = (u16*)carve(134217728);
  u16*   wbuf  = (u16*)carve(8388608);     // max 1024 x 3072 bf16
  float* stats = (float*)carve(16384);     // 2 x 2048 fp32
  float* l0xyz = (float*)carve(196608);
  float* l1xyz = (float*)carve(98304);
  float* l2xyz = (float*)carve(24576);
  float* l3xyz = (float*)carve(6144);
  float* l4xyz = (float*)carve(1536);
  int*   fidx  = (int*)carve(32768);
  int*   bidx  = (int*)carve(1048576);
  int*   nidx  = (int*)carve(196608);
  float* nwts  = (float*)carve(196608);
  float* l1p   = (float*)carve(8388608);   // (8,1024,256)
  float* l2p   = (float*)carve(4194304);   // (8,256,512)
  float* l3p   = (float*)carve(2097152);   // (8,64,1024)
  float* l4p   = (float*)carve(1048576);   // (8,16,2048)
  float* l3n   = (float*)carve(2097152);   // fp4 out
  float* l2n   = (float*)carve(4194304);   // fp3 out
  float* l1n   = (float*)carve(8388608);   // fp2 out
  float* l0n   = (float*)carve(8388608);   // fp1 out (8,2048,128)

  auto G = [](long tot) { return dim3((unsigned)((tot + 255) / 256)); };
  auto pad32 = [](int x) { return ((x + 31) / 32) * 32; };

  // ---- shared conv-chain runner (GEMM + BN/ReLU, bf16 ping-pong) ----
  auto run_convs = [&](const MlpP& m, const int* chans, int cin0, int lda0,
                       long rows, float* out_f32_last) -> u16* {
    u16* src = actA; u16* dst = actB;
    int cin = cin0, lda = lda0;
    for (int i = 0; i < m.n; ++i) {
      int cout = chans[i];
      long wtot = (long)cout * lda;
      k_convw<<<G(wtot), 256, 0, stream>>>(m.W[i], wbuf, cin, lda, wtot);
      k_zero<<<G(2 * cout), 256, 0, stream>>>(stats, 2 * cout);
      dim3 gg((unsigned)(rows / 256), (unsigned)(cout / 64));
      k_gemm_wmma<<<gg, 256, 0, stream>>>(src, wbuf, m.b[i], dst, stats,
                                          (int)rows, lda, cout);
      long tot = rows * (long)cout;
      float* of = (i == m.n - 1) ? out_f32_last : nullptr;
      k_bnrelu<<<G(tot), 256, 0, stream>>>(dst, stats, m.g[i], m.bt[i],
                                           1.0f / (float)rows, cout, tot, of);
      u16* t = src; src = dst; dst = t;
      cin = cout; lda = cout;
    }
    return src;  // final activation buffer
  };

  // ---- set-abstraction layer ----
  auto run_sa = [&](const float* xyzl, int Nl, const float* pts, int Cp,
                    int S, float radius, const MlpP& m, const int* chans,
                    float* nxyz, float* pooled) {
    k_fps<<<BATCH, 256, 0, stream>>>(xyzl, fidx, Nl, S);
    int q = BATCH * S;
    k_gather_xyz<<<G(q), 256, 0, stream>>>(xyzl, fidx, nxyz, Nl, S, q);
    k_ball<<<G(q), 256, 0, stream>>>(xyzl, nxyz, bidx, Nl, S, radius * radius, q);
    int lda0 = pad32(3 + Cp);
    long rows = (long)BATCH * S * KNB;
    long gt = rows * lda0;
    k_group<<<G(gt), 256, 0, stream>>>(xyzl, nxyz, bidx, pts, actA, Nl, S, Cp, lda0, gt);
    u16* fin = run_convs(m, chans, 3 + Cp, lda0, rows, nullptr);
    int Cl = chans[m.n - 1];
    long pt = (long)BATCH * S * Cl;
    k_maxpool<<<G(pt), 256, 0, stream>>>(fin, pooled, Cl, pt);
  };

  // ---- feature-propagation layer ----
  auto run_fp = [&](const float* xyz1, int N1, const float* xyz2, int N2,
                    const float* p1, int C1, const float* p2, int C2,
                    const MlpP& m, const int* chans, float* outf) {
    int q = BATCH * N1;
    k_knn3<<<G(q), 256, 0, stream>>>(xyz1, xyz2, nidx, nwts, N1, N2, q);
    int lda0 = pad32(C1 + C2);
    long rows = (long)BATCH * N1;
    long gt = rows * lda0;
    k_interp<<<G(gt), 256, 0, stream>>>(p1, p2, nidx, nwts, actA, N1, N2, C1, C2, lda0, gt);
    run_convs(m, chans, C1 + C2, lda0, rows, outf);
  };

  // ---- pipeline ----
  long t0 = (long)BATCH * NPTS0 * 3;
  k_tin<<<G(t0), 256, 0, stream>>>((const float*)d_in[0], l0xyz, NPTS0, t0);

  run_sa(l0xyz, 2048, nullptr, 0,   1024, 0.1f, sa1, sa1c, l1xyz, l1p);
  run_sa(l1xyz, 1024, l1p,     256, 256,  0.2f, sa2, sa2c, l2xyz, l2p);
  run_sa(l2xyz, 256,  l2p,     512, 64,   0.4f, sa3, sa3c, l3xyz, l3p);
  run_sa(l3xyz, 64,   l3p,     1024, 16,  0.8f, sa4, sa4c, l4xyz, l4p);

  run_fp(l3xyz, 64,   l4xyz, 16,   l3p,    1024, l4p, 2048, fp4, fp4c, l3n);
  run_fp(l2xyz, 256,  l3xyz, 64,   l2p,    512,  l3n, 1024, fp3, fp3c, l2n);
  run_fp(l1xyz, 1024, l2xyz, 256,  l1p,    256,  l2n, 512,  fp2, fp2c, l1n);
  run_fp(l0xyz, 2048, l1xyz, 1024, nullptr, 0,   l1n, 256,  fp1, fp1c, l0n);

  // ---- outputs: (B,N,C) -> (B,C,N), concatenated ----
  float* out = (float*)d_out;
  long s0 = (long)BATCH * 128 * 2048;
  long s1 = (long)BATCH * 256 * 1024;
  long s2 = (long)BATCH * 512 * 256;
  long s3 = (long)BATCH * 1024 * 64;
  k_tout<<<G(s0), 256, 0, stream>>>(l0n, out,                2048, 128,  s0);
  k_tout<<<G(s1), 256, 0, stream>>>(l1n, out + s0,           1024, 256,  s1);
  k_tout<<<G(s2), 256, 0, stream>>>(l2n, out + s0 + s1,      256,  512,  s2);
  k_tout<<<G(s3), 256, 0, stream>>>(l3n, out + s0 + s1 + s2, 64,   1024, s3);
}
